// AttentiveStatisticsPooling_3178275799424
// MI455X (gfx1250) — matrix operand order repaired and checked by reading, not compiled, not run on hardware
//
#include <hip/hip_runtime.h>
#include <hip/hip_bf16.h>

// Problem constants (fixed by the reference setup)
#define B_  32
#define C_  512
#define T_  2000
#define D_  128
#define C3_ 1536
#define NT_ (T_ / 16)   // 125 t-tiles
#define EPSV 1e-8f
#define NEGV -1e9f

typedef __attribute__((ext_vector_type(16))) _Float16 v16h;
typedef __attribute__((ext_vector_type(8)))  float    v8f;
typedef __attribute__((ext_vector_type(4)))  unsigned int v4u;
typedef __attribute__((ext_vector_type(4)))  unsigned int u32x4;
typedef __attribute__((ext_vector_type(8)))  int          i32x8;
typedef __attribute__((ext_vector_type(4)))  int          i32x4;

// ---------------------------------------------------------------------------
// Kernel 0: convert W1 (x-part of attn_w, D x C) and ctx_w (C x D) to f16
// ---------------------------------------------------------------------------
__global__ __launch_bounds__(256) void convert_kernel(const float* __restrict__ attn_w,
                                                      const float* __restrict__ ctx_w,
                                                      _Float16* __restrict__ w1h,
                                                      _Float16* __restrict__ cwh) {
    int i = blockIdx.x * 256 + threadIdx.x;
    if (i < D_ * C_) {
        int d = i >> 9, c = i & (C_ - 1);
        w1h[i] = (_Float16)attn_w[(size_t)d * C3_ + c]; // W1 = attn_w[:, 0:C]
        cwh[i] = (_Float16)ctx_w[i];                    // (C, D) row-major as-is
    }
}

// ---------------------------------------------------------------------------
// Kernel 1: masked mean/std per (b, c) over T
// ---------------------------------------------------------------------------
__global__ __launch_bounds__(256) void stats_kernel(const float* __restrict__ x,
                                                    const int* __restrict__ mask,
                                                    float* __restrict__ mean,
                                                    float* __restrict__ stdv) {
    int bc = blockIdx.x;            // b*C + c
    int b  = bc >> 9;               // / C_
    const float* xr = x + (size_t)bc * T_;
    const int*   mr = mask + (size_t)b * T_;
    float sm = 0.f, sq = 0.f, cnt = 0.f;
    for (int t = threadIdx.x; t < T_; t += 256) {
        float mv = (float)mr[t];
        float xv = xr[t];
        sm = fmaf(mv, xv, sm);
        sq = fmaf(mv * xv, xv, sq);
        cnt += mv;
    }
    for (int off = 1; off < 32; off <<= 1) {
        sm  += __shfl_xor(sm,  off, 32);
        sq  += __shfl_xor(sq,  off, 32);
        cnt += __shfl_xor(cnt, off, 32);
    }
    __shared__ float r0[8], r1[8], r2[8];
    int w = threadIdx.x >> 5;
    if ((threadIdx.x & 31) == 0) { r0[w] = sm; r1[w] = sq; r2[w] = cnt; }
    __syncthreads();
    if (threadIdx.x == 0) {
        float S = 0.f, Q = 0.f, L = 0.f;
        for (int i = 0; i < 8; ++i) { S += r0[i]; Q += r1[i]; L += r2[i]; }
        float len = fmaxf(L, 1.0f);
        float mu  = S / len;
        float var = Q / len - mu * mu;
        mean[bc] = mu;
        stdv[bc] = sqrtf(fmaxf(var, EPSV));
    }
}

// ---------------------------------------------------------------------------
// Kernel 1b: beff[b][d] = attn_b[d] + sum_c W2[d][c]*mean[b][c] + W3[d][c]*std[b][c]
// One wave (32 lanes) per (b, d) dot product.
// ---------------------------------------------------------------------------
__global__ __launch_bounds__(128) void beff_kernel(const float* __restrict__ attn_w,
                                                   const float* __restrict__ attn_b,
                                                   const float* __restrict__ mean,
                                                   const float* __restrict__ stdv,
                                                   float* __restrict__ beff) {
    int gid  = blockIdx.x * 4 + (threadIdx.x >> 5);   // (b, d) pair
    int lane = threadIdx.x & 31;
    int b = gid >> 7;            // / D_
    int d = gid & (D_ - 1);
    const float* w2 = attn_w + (size_t)d * C3_ + C_;
    const float* w3 = w2 + C_;
    const float* mb = mean + (size_t)b * C_;
    const float* sb = stdv + (size_t)b * C_;
    float acc = 0.f;
    for (int c = lane; c < C_; c += 32)
        acc += w2[c] * mb[c] + w3[c] * sb[c];
    for (int off = 1; off < 32; off <<= 1) acc += __shfl_xor(acc, off, 32);
    if (lane == 0) beff[gid] = acc + attn_b[d];
}

// ---------------------------------------------------------------------------
// Kernel 2: s[b][t][d] = tanh( sum_c W1[d][c] * x[b][c][t] + beff[b][d] ) as f16
// One workgroup = (b, 16-column t-tile); 4 waves, each wave owns 2 m-tiles of D.
//
// Pipeline:
//  1) TENSOR_LOAD_TO_LDS (Tensor Data Mover): ONE instruction DMAs the whole
//     16x512 f32 tile (rows of 64 B, stride T*4) into LDS, with the TDM's
//     hardware padding (4 dwords after every 16) producing the XSF_STRIDE=20
//     layout directly. Tracked by TENSORcnt; waited via s_wait_tensorcnt.
//  2) One f32->f16 convert+transpose step into xsh[t][c-pairs] (once per
//     block, NOT once per wave) so the hot loop has zero converts.
//  3) Hot loop: B fragment = 2x ds_load_b128 (pair-dwords are contiguous),
//     A fragments = global_load_b128 of W1 f16, 2 WMMAs per k-step.
// ---------------------------------------------------------------------------
#define XSF_STRIDE 20    // dwords per 16-float x row (16 data + 4 TDM pad)
#define XSH_STRIDE 260   // dwords per t-row of 256 f16-pairs (multiple of 4 for b128)
__global__ __launch_bounds__(128) void gemm1_tanh_kernel(const _Float16* __restrict__ w1h,
                                                         const float* __restrict__ x,
                                                         const float* __restrict__ beff,
                                                         _Float16* __restrict__ sbuf) {
    __shared__ float        xsf[C_ * XSF_STRIDE];   // 40 KB: raw f32 x rows
    __shared__ unsigned int xsh[16 * XSH_STRIDE];   // 16.6 KB: f16 pairs, [t][c]
    const int t0  = blockIdx.x * 16;
    const int b   = blockIdx.y;
    const int tid = threadIdx.x;

    // (1) TDM: x[0:512][t0:t0+16] -> xsf, one descriptor, issued by wave 0.
    if (tid < 32) {
        unsigned long long gaddr =
            (unsigned long long)(size_t)(x + (size_t)b * C_ * T_ + t0);
        unsigned int ldsa = (unsigned int)(size_t)(&xsf[0]);  // flat low bits = LDS offset
        // D# group 0: count=1 (valid), lds_addr, global_addr[56:0], type=2 ("image")
        u32x4 g0;
        g0.x = 1u;
        g0.y = ldsa;
        g0.z = (unsigned int)gaddr;
        g0.w = ((unsigned int)(gaddr >> 32) & 0x01FFFFFFu) | 0x80000000u;
        // D# group 1: data_size=2 (4B), pad_enable=1, pad_interval=3 (every 16
        // dwords), pad_amount=3 (4 dwords); tensor_dim0=16, tensor_dim1=512;
        // tile_dim0=16, tile_dim1=512; tensor_dim0_stride=T elements.
        i32x8 g1;
        g1[0] = (2 << 16) | (1 << 20) | (3 << 22) | (3 << 25);
        g1[1] = (16 << 16);          // tensor_dim0[15:0] in bits[63:48]
        g1[2] = (512 << 16);         // tensor_dim1[15:0] in bits[95:80]
        g1[3] = (16 << 16);          // tile_dim0 in bits[127:112]
        g1[4] = 512;                 // tile_dim1 in bits[143:128]
        g1[5] = T_;                  // tensor_dim0_stride[31:0]
        g1[6] = 0;
        g1[7] = 0;
        i32x4 gz  = {0, 0, 0, 0};              // groups 2/3 unused (2D tensor)
        i32x8 gz8 = {0, 0, 0, 0, 0, 0, 0, 0};  // 6-arg form extra group (unused)
        __builtin_amdgcn_tensor_load_to_lds(g0, g1, gz, gz, gz8, 0);
        __builtin_amdgcn_s_wait_tensorcnt(0);
    }
    __syncthreads();

    // (2) convert + transpose: xsh[t][cp] = pack(f16(x[2cp][t]), f16(x[2cp+1][t]))
    {
        int tl = tid & 15;
        for (int cp = (tid >> 4); cp < C_ / 2; cp += 8) {
            float lo = xsf[(2 * cp)     * XSF_STRIDE + tl];
            float hi = xsf[(2 * cp + 1) * XSF_STRIDE + tl];
            union { unsigned int u; _Float16 h[2]; } p;
            p.h[0] = (_Float16)lo;
            p.h[1] = (_Float16)hi;
            xsh[tl * XSH_STRIDE + cp] = p.u;
        }
    }
    __syncthreads();

    const int lane = tid & 31;
    const int w    = tid >> 5;
    const int hi   = lane >> 4;       // 0: lanes 0-15, 1: lanes 16-31
    const int mrow = lane & 15;       // A-matrix row M
    const int n    = lane & 15;       // B/C/D column N

    v8f acc0 = {};
    v8f acc1 = {};
    const int d0 = (2 * w) * 16;      // first m-tile row base
    const int d1 = d0 + 16;           // second m-tile row base

    // (3) hot loop over K = C in steps of 32
    for (int k0 = 0; k0 < C_; k0 += 32) {
        // B fragment (32x16 f16): 8 contiguous pair-dwords -> two ds_load_b128
        union { v16h v; v4u q[2]; } bf;
        const unsigned int* bp = &xsh[n * XSH_STRIDE + (k0 >> 1) + 8 * hi];
        bf.q[0] = *(const v4u*)(bp);
        bf.q[1] = *(const v4u*)(bp + 4);
        // A fragments (16x32 f16): lane holds row mrow; VGPR v packs K pair
        union { v16h v; unsigned int u[8]; } a0, a1;
#pragma unroll
        for (int v = 0; v < 8; ++v) {
            int kb = k0 + 2 * v + (hi ? 16 : 0);
            a0.u[v] = *(const unsigned int*)(w1h + (size_t)(d0 + mrow) * C_ + kb);
            a1.u[v] = *(const unsigned int*)(w1h + (size_t)(d1 + mrow) * C_ + kb);
        }
        acc0 = __builtin_amdgcn_wmma_f32_16x16x32_f16(false, a0.v, false, bf.v,
                                                      (short)0, acc0, false, false);
        acc1 = __builtin_amdgcn_wmma_f32_16x16x32_f16(false, a1.v, false, bf.v,
                                                      (short)0, acc1, false, false);
    }

    // Epilogue: bias + tanh, store transposed f16 to s[b][(t0+n)*D + d]
    const float* be = beff + (size_t)b * D_;
    _Float16* srow = sbuf + ((size_t)b * T_ + (t0 + n)) * D_;
#pragma unroll
    for (int r = 0; r < 8; ++r) {
        int m0 = r + 8 * hi;          // C/D fragment row
        float v0 = tanhf(acc0[r] + be[d0 + m0]);
        float v1 = tanhf(acc1[r] + be[d1 + m0]);
        srow[d0 + m0] = (_Float16)v0;
        srow[d1 + m0] = (_Float16)v1;
    }
}

// ---------------------------------------------------------------------------
// Kernel 3: fused GEMM2 + masked softmax + attentive pooling.
// One workgroup = (b, 32-row c-tile): 2 m-tiles per wave (halves the number of
// passes over the L2-resident activation slab s vs a 16-row tile). 4 waves
// split t-tiles. Pass A: WMMA logits, track per-row max only. Pass B:
// recompute logits, accumulate sum(e), sum(e*x), sum(e*x^2).
// var = E[x^2] - E[x]^2. Never materializes the 131 MB (B,C,T) alpha tensor.
// ---------------------------------------------------------------------------
__global__ __launch_bounds__(128) void attn_pool_kernel(const _Float16* __restrict__ cwh,
                                                        const _Float16* __restrict__ sbuf,
                                                        const float* __restrict__ ctx_b,
                                                        const float* __restrict__ x,
                                                        const int* __restrict__ mask,
                                                        float* __restrict__ out) {
    const int c0   = blockIdx.x * 32;
    const int b    = blockIdx.y;
    const int tid  = threadIdx.x;
    const int lane = tid & 31;
    const int w    = tid >> 5;
    const int hi   = lane >> 4;
    const int mrow = lane & 15;
    const int n    = lane & 15;

    // Preload A fragments for 2 m-tiles x 4 K-steps (K = D = 128), ctx_w f16
    union AF { v16h v; unsigned int u[8]; } af[2][4];
#pragma unroll
    for (int mt = 0; mt < 2; ++mt)
#pragma unroll
        for (int ks = 0; ks < 4; ++ks)
#pragma unroll
            for (int v = 0; v < 8; ++v) {
                int kb = ks * 32 + 2 * v + (hi ? 16 : 0);
                af[mt][ks].u[v] = *(const unsigned int*)(
                    cwh + (size_t)(c0 + 16 * mt + mrow) * D_ + kb);
            }

    float cb[2][8];
#pragma unroll
    for (int mt = 0; mt < 2; ++mt)
#pragma unroll
        for (int r = 0; r < 8; ++r)
            cb[mt][r] = ctx_b[c0 + 16 * mt + r + 8 * hi];

    const _Float16* sb = sbuf + (size_t)b * T_ * D_;
    const int* mk = mask + (size_t)b * T_;

    // ---- Pass A: per-row max of logits over this wave's t-tiles ----
    float rmax[2][8];
#pragma unroll
    for (int mt = 0; mt < 2; ++mt)
#pragma unroll
        for (int r = 0; r < 8; ++r) rmax[mt][r] = -3.0e38f;

    for (int tt = w; tt < NT_; tt += 4) {
        int t0 = tt * 16;
        __builtin_prefetch(sb + (size_t)(t0 + 64 + n) * D_, 0, 0);
        union BF { v16h v; unsigned int u[8]; } bf;
        v8f acc0 = {};
        v8f acc1 = {};
#pragma unroll
        for (int ks = 0; ks < 4; ++ks) {
#pragma unroll
            for (int v = 0; v < 8; ++v) {
                int kb = ks * 32 + 2 * v + (hi ? 16 : 0);
                bf.u[v] = *(const unsigned int*)(sb + (size_t)(t0 + n) * D_ + kb);
            }
            acc0 = __builtin_amdgcn_wmma_f32_16x16x32_f16(false, af[0][ks].v, false, bf.v,
                                                          (short)0, acc0, false, false);
            acc1 = __builtin_amdgcn_wmma_f32_16x16x32_f16(false, af[1][ks].v, false, bf.v,
                                                          (short)0, acc1, false, false);
        }
        int mv = mk[t0 + n];
#pragma unroll
        for (int r = 0; r < 8; ++r) {
            float lg0 = acc0[r] + cb[0][r];
            float lg1 = acc1[r] + cb[1][r];
            if (mv == 0) { lg0 = NEGV; lg1 = NEGV; }
            rmax[0][r] = fmaxf(rmax[0][r], lg0);
            rmax[1][r] = fmaxf(rmax[1][r], lg1);
        }
    }
    // reduce max within the 16-lane group holding each row
#pragma unroll
    for (int mt = 0; mt < 2; ++mt)
#pragma unroll
        for (int r = 0; r < 8; ++r)
            for (int off = 1; off < 16; off <<= 1)
                rmax[mt][r] = fmaxf(rmax[mt][r], __shfl_xor(rmax[mt][r], off, 32));

    __shared__ float smax[4][32];
    __shared__ float spart[4][32][3];
    if ((lane & 15) < 8) {
#pragma unroll
        for (int mt = 0; mt < 2; ++mt)
            smax[w][16 * mt + (lane & 7) + 8 * hi] = rmax[mt][lane & 7];
    }
    __syncthreads();

    float gmax[2][8];
#pragma unroll
    for (int mt = 0; mt < 2; ++mt)
#pragma unroll
        for (int r = 0; r < 8; ++r) {
            int row = 16 * mt + r + 8 * hi;
            gmax[mt][r] = fmaxf(fmaxf(smax[0][row], smax[1][row]),
                                fmaxf(smax[2][row], smax[3][row]));
        }

    // ---- Pass B: recompute logits, accumulate softmax-weighted stats ----
    float se[2][8], sm[2][8], sq[2][8];
#pragma unroll
    for (int mt = 0; mt < 2; ++mt)
#pragma unroll
        for (int r = 0; r < 8; ++r) { se[mt][r] = 0.f; sm[mt][r] = 0.f; sq[mt][r] = 0.f; }

    for (int tt = w; tt < NT_; tt += 4) {
        int t0 = tt * 16;
        __builtin_prefetch(sb + (size_t)(t0 + 64 + n) * D_, 0, 0);
        union BF { v16h v; unsigned int u[8]; } bf;
        v8f acc0 = {};
        v8f acc1 = {};
#pragma unroll
        for (int ks = 0; ks < 4; ++ks) {
#pragma unroll
            for (int v = 0; v < 8; ++v) {
                int kb = ks * 32 + 2 * v + (hi ? 16 : 0);
                bf.u[v] = *(const unsigned int*)(sb + (size_t)(t0 + n) * D_ + kb);
            }
            acc0 = __builtin_amdgcn_wmma_f32_16x16x32_f16(false, af[0][ks].v, false, bf.v,
                                                          (short)0, acc0, false, false);
            acc1 = __builtin_amdgcn_wmma_f32_16x16x32_f16(false, af[1][ks].v, false, bf.v,
                                                          (short)0, acc1, false, false);
        }
        int mv = mk[t0 + n];
#pragma unroll
        for (int mt = 0; mt < 2; ++mt) {
#pragma unroll
            for (int r = 0; r < 8; ++r) {
                float lg = (mt == 0 ? acc0[r] : acc1[r]) + cb[mt][r];
                if (mv == 0) lg = NEGV;
                float e  = __expf(lg - gmax[mt][r]);
                float xv = x[((size_t)b * C_ + (c0 + 16 * mt + r + 8 * hi)) * T_ + t0 + n];
                se[mt][r] += e;
                sm[mt][r] = fmaf(e, xv, sm[mt][r]);
                sq[mt][r] = fmaf(e * xv, xv, sq[mt][r]);
            }
        }
    }
    // reduce within 16-lane group
#pragma unroll
    for (int mt = 0; mt < 2; ++mt)
#pragma unroll
        for (int r = 0; r < 8; ++r)
            for (int off = 1; off < 16; off <<= 1) {
                se[mt][r] += __shfl_xor(se[mt][r], off, 32);
                sm[mt][r] += __shfl_xor(sm[mt][r], off, 32);
                sq[mt][r] += __shfl_xor(sq[mt][r], off, 32);
            }
    if ((lane & 15) < 8) {
#pragma unroll
        for (int mt = 0; mt < 2; ++mt) {
            int row = 16 * mt + (lane & 7) + 8 * hi;
            spart[w][row][0] = se[mt][lane & 7];
            spart[w][row][1] = sm[mt][lane & 7];
            spart[w][row][2] = sq[mt][lane & 7];
        }
    }
    __syncthreads();

    if (tid < 32) {
        float SE = 0.f, SM = 0.f, SQ = 0.f;
        for (int ww = 0; ww < 4; ++ww) {
            SE += spart[ww][tid][0];
            SM += spart[ww][tid][1];
            SQ += spart[ww][tid][2];
        }
        float pm  = SM / SE;
        float var = SQ / SE - pm * pm;
        float ps  = sqrtf(fmaxf(var, EPSV));
        out[(size_t)b * (2 * C_) + c0 + tid]      = pm;   // pooled mean
        out[(size_t)b * (2 * C_) + C_ + c0 + tid] = ps;   // pooled std
    }
}

// ---------------------------------------------------------------------------
// Host launcher
// ---------------------------------------------------------------------------
extern "C" void kernel_launch(void* const* d_in, const int* in_sizes, int n_in,
                              void* d_out, int out_size, void* d_ws, size_t ws_size,
                              hipStream_t stream) {
    const float* x      = (const float*)d_in[0];   // (B, C, T)
    const float* attn_w = (const float*)d_in[1];   // (D, 3C)
    const float* attn_b = (const float*)d_in[2];   // (D,)
    const float* ctx_w  = (const float*)d_in[3];   // (C, D)
    const float* ctx_b  = (const float*)d_in[4];   // (C,)
    const int*   mask   = (const int*)d_in[5];     // (B, T)
    float* out = (float*)d_out;                    // (B, 2C)

    // Workspace carve-up (~16.8 MB)
    float* mean = (float*)d_ws;                    // B*C
    float* stdv = mean + B_ * C_;                  // B*C
    float* beff = stdv + B_ * C_;                  // B*D
    _Float16* w1h  = (_Float16*)(beff + B_ * D_);  // D*C f16
    _Float16* cwh  = w1h + D_ * C_;                // C*D f16
    _Float16* sbuf = cwh + C_ * D_;                // B*T*D f16 (tanh acts, transposed)

    convert_kernel<<<(D_ * C_ + 255) / 256, 256, 0, stream>>>(attn_w, ctx_w, w1h, cwh);
    stats_kernel<<<B_ * C_, 256, 0, stream>>>(x, mask, mean, stdv);
    beff_kernel<<<(B_ * D_) / 4, 128, 0, stream>>>(attn_w, attn_b, mean, stdv, beff);
    gemm1_tanh_kernel<<<dim3(NT_, B_), 128, 0, stream>>>(w1h, x, beff, sbuf);
    attn_pool_kernel<<<dim3(C_ / 32, B_), 128, 0, stream>>>(cwh, sbuf, ctx_b, x, mask, out);
}
